// E3nnModel_84172769067731
// MI455X (gfx1250) — matrix operand order repaired and checked by reading, not compiled, not run on hardware
//
#include <hip/hip_runtime.h>
#include <math.h>

typedef __attribute__((ext_vector_type(16))) _Float16     v16h;
typedef __attribute__((ext_vector_type(8)))  float        v8f;
typedef __attribute__((ext_vector_type(8)))  unsigned int v8u;

#define IN_DIM   118
#define EM       32
#define NBASIS   10
#define HID      100
#define HIDP     112     // HID padded to multiple of 16 (GEMM1 output cols)
#define HIDP2    128     // K padding for GEMM2 / LDS row stride in halves
#define OUT_MUL  16
#define MAX_R    5.0f

// ---------------------------------------------------------------------------
// WMMA helpers (layouts per CDNA5 ISA 7.12.2, wave32)
// ---------------------------------------------------------------------------
__device__ __forceinline__ v8f wmma_f16(v16h a, v16h b, v8f c) {
  return __builtin_amdgcn_wmma_f32_16x16x32_f16(false, a, false, b, (short)0, c,
                                                false, false);
}

// Packed B layout: pb[kblock * ncols + col] is a v16h holding
// W[kblock*16 + 0..15][col].  Lane (col, hi) of a wave reads one 32B vector:
// two global_load_b128, coalesced across lanes.
__device__ __forceinline__ v16h load_bp(const v16h* __restrict__ pb, int ncols,
                                        int kbase, int col, int hi) {
  return pb[((kbase >> 4) + hi) * ncols + col];
}

// A-matrix (16 x 32, 16-bit): within-chunk K offset for element i of a lane
__device__ __forceinline__ int jmapA(int i, int hi) {
  return i + (hi ? 16 : 0);
}

// ---------------------------------------------------------------------------
// Kw: convert weights to f16 scratch copies, pre-swizzled into WMMA B layout
// ---------------------------------------------------------------------------
__global__ void k_convert(const float* Wl1, const float* Wsc, const float* Wl2s,
                          const float* Wl2v, const float* Wfc1, const float* Wfc2,
                          _Float16* wl1p, _Float16* wscp, _Float16* wl2sp,
                          _Float16* wl2vp, _Float16* wfc1p, _Float16* wfc2p) {
  int i = blockIdx.x * blockDim.x + threadIdx.x;
  const int n1 = 1024 * 32;        // W_lin1  (K=1024, ncols=32)
  const int n2 = n1 + 1024 * 16;   // W_sc    (K=1024, ncols=16)
  const int n3 = n2 + 1024 * 16;   // W_lin2_s
  const int n4 = n3 + 1024 * 16;   // W_lin2_v
  const int n5 = n4 + 32 * HIDP;   // W_fc1 padded (K=32, ncols=112)
  const int n6 = n5 + 128 * 64;    // W_fc2 padded (K=128, ncols=64)
  if (i < n1) {
    int p = i;                       // packed idx: kb*32*16 + col*16 + ii
    int kb = p >> 9, rem = p & 511, col = rem >> 4, ii = rem & 15;
    wl1p[p] = (_Float16)Wl1[(kb * 16 + ii) * 32 + col];
  } else if (i < n2) {
    int p = i - n1;
    int kb = p >> 8, rem = p & 255, col = rem >> 4, ii = rem & 15;
    wscp[p] = (_Float16)Wsc[(kb * 16 + ii) * 16 + col];
  } else if (i < n3) {
    int p = i - n2;
    int kb = p >> 8, rem = p & 255, col = rem >> 4, ii = rem & 15;
    wl2sp[p] = (_Float16)Wl2s[(kb * 16 + ii) * 16 + col];
  } else if (i < n4) {
    int p = i - n3;
    int kb = p >> 8, rem = p & 255, col = rem >> 4, ii = rem & 15;
    wl2vp[p] = (_Float16)Wl2v[(kb * 16 + ii) * 16 + col];
  } else if (i < n5) {
    int p = i - n4;
    int kb = p / (HIDP * 16), rem = p % (HIDP * 16), col = rem >> 4, ii = rem & 15;
    int k = kb * 16 + ii;
    wfc1p[p] = (_Float16)((k < NBASIS && col < HID) ? Wfc1[k * HID + col] : 0.0f);
  } else if (i < n6) {
    int p = i - n5;
    int kb = p >> 10, rem = p & 1023, col = rem >> 4, ii = rem & 15;
    int k = kb * 16 + ii;
    wfc2p[p] = (_Float16)((k < HID) ? Wfc2[k * 64 + col] : 0.0f);
  }
}

// ---------------------------------------------------------------------------
// K0: shared embedding x0 = relu(in@W+b), a = relu(attr@W+b); graph counts
// ---------------------------------------------------------------------------
__global__ void k_embed(const float* __restrict__ node_input,
                        const float* __restrict__ node_attr,
                        const float* __restrict__ W_em, const float* __restrict__ b_em,
                        const int* __restrict__ batch,
                        float* __restrict__ x0, float* __restrict__ a,
                        float* __restrict__ cnt, int N) {
  int gid = blockIdx.x * blockDim.x + threadIdx.x;
  if (gid >= N * EM) return;
  int n = gid >> 5, c = gid & 31;
  float s0 = 0.f, s1 = 0.f;
  const float* ri = node_input + (size_t)n * IN_DIM;
  const float* ra = node_attr + (size_t)n * IN_DIM;
#pragma unroll 2
  for (int k = 0; k < IN_DIM; ++k) {
    float w = W_em[k * EM + c];
    s0 += ri[k] * w;
    s1 += ra[k] * w;
  }
  float b = b_em[c];
  x0[gid] = fmaxf(s0 + b, 0.f);
  a[gid]  = fmaxf(s1 + b, 0.f);
  if (c == 0) atomicAdd(&cnt[batch[n]], 1.0f);
}

// ---------------------------------------------------------------------------
// K1: FCTP over nodes: x1 = einsum(x0,a,W_lin1)/32 ; s_sc = einsum(x0,a,W_sc)/32
// One wave per 16-node tile; K = 1024 = 32 chunks of rank-1 A tiles.
// A build is pure packed-f16: v16h * splatted scalar -> 8x v_pk_mul_f16.
// ---------------------------------------------------------------------------
__global__ void __launch_bounds__(32)
k_fctp1(const float* __restrict__ x0, const float* __restrict__ a,
        const v16h* __restrict__ wl1p, const v16h* __restrict__ wscp,
        float* __restrict__ x1, float* __restrict__ ssc, int N) {
  __shared__ float x0s[16 * EM];
  __shared__ float as[16 * EM];
  const int base = blockIdx.x * 16;
  const int lane = threadIdx.x;
  const int row = lane & 15, col = lane & 15;
  const int hi = lane >> 4;
  // vectorized tile load: 8 float4 per row
#pragma unroll
  for (int i = 0; i < 4; ++i) {
    int f4 = lane + 32 * i;            // float4 index within the 16x32 tile
    int n = base + (f4 >> 3), c4 = (f4 & 7) * 4;
    float4 vx = make_float4(0.f, 0.f, 0.f, 0.f), va = vx;
    if (n < N) {
      vx = *(const float4*)(x0 + (size_t)n * EM + c4);
      va = *(const float4*)(a + (size_t)n * EM + c4);
    }
    *(float4*)(x0s + f4 * 4) = vx;
    *(float4*)(as + f4 * 4) = va;
  }
  __syncthreads();
  v16h av16;
#pragma unroll
  for (int i = 0; i < 16; ++i) av16[i] = (_Float16)as[row * EM + jmapA(i, hi)];

  v8f accA = {}, accB = {}, accC = {};
#pragma unroll 2
  for (int c = 0; c < 32; ++c) {
    _Float16 xh = (_Float16)x0s[row * EM + c];
    v16h A = av16 * xh;                       // 8x v_pk_mul_f16
    v16h B0 = load_bp(wl1p, 32, 32 * c, col, hi);
    v16h B1 = load_bp(wl1p, 32, 32 * c, col + 16, hi);
    v16h B2 = load_bp(wscp, 16, 32 * c, col, hi);
    accA = wmma_f16(A, B0, accA);
    accB = wmma_f16(A, B1, accB);
    accC = wmma_f16(A, B2, accC);
  }
  const float inv = 1.0f / 32.0f;
#pragma unroll
  for (int j = 0; j < 8; ++j) {
    int M = j + (hi ? 8 : 0);
    int n = base + M;
    if (n < N) {
      x1[(size_t)n * EM + col]       = accA[j] * inv;
      x1[(size_t)n * EM + col + 16]  = accB[j] * inv;
      ssc[(size_t)n * OUT_MUL + col] = accC[j] * inv;
    }
  }
}

// ---------------------------------------------------------------------------
// K2: per-edge: geometry -> radial MLP (WMMA) -> message -> atomic scatter.
// One wave per 16-edge tile.  h is staged in LDS as f16 (row stride 128,
// pad zeroed) so the GEMM2 A tile is 8 aligned ds_load_b32, no conversions.
// ---------------------------------------------------------------------------
__global__ void __launch_bounds__(32)
k_edge(const int* __restrict__ edge_index, const float* __restrict__ edge_vec,
       const float* __restrict__ x1,
       const v16h* __restrict__ wfc1p, const v16h* __restrict__ wfc2p,
       float* __restrict__ aggs, float* __restrict__ aggv, int E) {
  __shared__ _Float16 hlds[16 * HIDP2];
  __shared__ float ea[16 * 4];
  __shared__ int sd[16 * 2];
  const int e0 = blockIdx.x * 16;
  const int lane = threadIdx.x;
  const int row = lane & 15, col = lane & 15;
  const int hi = lane >> 4;
  const int e = e0 + row;
  const bool ev = (e < E);

  float vx = 0.f, vy = 0.f, vz = 0.f;
  if (ev) {
    vx = edge_vec[3 * (size_t)e];
    vy = edge_vec[3 * (size_t)e + 1];
    vz = edge_vec[3 * (size_t)e + 2];
  }
  float r = sqrtf(vx * vx + vy * vy + vz * vz);
  float rin = 1.0f / fmaxf(r, 1e-6f);
  const float s3 = 1.7320508075688772f;
  float shx = s3 * vx * rin, shy = s3 * vy * rin, shz = s3 * vz * rin;
  float u = 2.0f * (r / MAX_R - 1.0f);
  float cut = (u > 0.f) ? 0.f : ((u < -1.f) ? 1.f : 0.5f * (1.0f - cosf(3.14159265358979f * u)));
  if (lane < 16) {
    ea[row * 4 + 0] = cut;
    ea[row * 4 + 1] = cut * shx;
    ea[row * 4 + 2] = cut * shy;
    ea[row * 4 + 3] = cut * shz;
    sd[row * 2 + 0] = ev ? edge_index[e] : 0;
    sd[row * 2 + 1] = ev ? edge_index[(size_t)E + e] : 0;
    // zero the K pad region (cols 112..127) of this row
#pragma unroll
    for (int p = 0; p < 8; ++p)
      *(unsigned int*)(hlds + row * HIDP2 + HIDP + 2 * p) = 0u;
  }
  // Gaussian soft one-hot in WMMA A layout (K padded 10 -> 32)
  const float step = MAX_R / (NBASIS - 1);
  const float embscale = 2.8231749f;   // sqrt(10)/1.12
  v16h Ae;
#pragma unroll
  for (int i = 0; i < 16; ++i) {
    int j = jmapA(i, hi);
    float val = 0.f;
    if (j < NBASIS && ev) {
      float t = (r - step * (float)j) / step;
      val = expf(-t * t) * embscale;
    }
    Ae[i] = (_Float16)val;
  }
  // GEMM1: emb @ W_fc1, then silu -> LDS (f16)
  const float s1 = 0.31622776601f;  // 1/sqrt(NB)
#pragma unroll
  for (int t = 0; t < 7; ++t) {
    v16h B = load_bp(wfc1p, HIDP, 0, 16 * t + col, hi);
    v8f acc = {};
    acc = wmma_f16(Ae, B, acc);
#pragma unroll
    for (int j = 0; j < 8; ++j) {
      int M = j + (hi ? 8 : 0);
      float p = acc[j] * s1;
      hlds[M * HIDP2 + 16 * t + col] = (_Float16)(p / (1.0f + expf(-p)));
    }
  }
  __syncthreads();
  // GEMM2: h @ W_fc2 (K padded 100 -> 128), 4 output col-tiles of 16
  v8f w0 = {}, w1 = {}, w2 = {}, w3 = {};
#pragma unroll
  for (int c = 0; c < 4; ++c) {
    v8u Au;
#pragma unroll
    for (int i2 = 0; i2 < 8; ++i2) {
      int koff = 2 * i2 + (hi ? 16 : 0);
      Au[i2] = *(const unsigned int*)(hlds + row * HIDP2 + 32 * c + koff);
    }
    v16h Ah = __builtin_bit_cast(v16h, Au);
    w0 = wmma_f16(Ah, load_bp(wfc2p, 64, 32 * c, col, hi), w0);
    w1 = wmma_f16(Ah, load_bp(wfc2p, 64, 32 * c, col + 16, hi), w1);
    w2 = wmma_f16(Ah, load_bp(wfc2p, 64, 32 * c, col + 32, hi), w2);
    w3 = wmma_f16(Ah, load_bp(wfc2p, 64, 32 * c, col + 48, hi), w3);
  }
  // Message + scatter-add (agg arrays are L2-resident; non-returning atomics)
  const float s2 = 0.1f;  // 1/sqrt(HID)
#pragma unroll
  for (int j = 0; j < 8; ++j) {
    int M = j + (hi ? 8 : 0);
    int ee = e0 + M;
    if (ee >= E) continue;
    int sn = sd[M * 2 + 0], dn = sd[M * 2 + 1];
    float a0 = ea[M * 4 + 0], a1 = ea[M * 4 + 1], a2 = ea[M * 4 + 2], a3 = ea[M * 4 + 3];
    float xg0 = x1[(size_t)sn * EM + col];
    float xg1 = x1[(size_t)sn * EM + col + 16];
    atomicAdd(&aggs[(size_t)dn * EM + col],      w0[j] * s2 * xg0 * a0);
    atomicAdd(&aggs[(size_t)dn * EM + col + 16], w1[j] * s2 * xg1 * a0);
    float fv0 = w2[j] * s2 * xg0;
    atomicAdd(&aggv[(size_t)dn * 96 + col * 3 + 0], fv0 * a1);
    atomicAdd(&aggv[(size_t)dn * 96 + col * 3 + 1], fv0 * a2);
    atomicAdd(&aggv[(size_t)dn * 96 + col * 3 + 2], fv0 * a3);
    float fv1 = w3[j] * s2 * xg1;
    atomicAdd(&aggv[(size_t)dn * 96 + (col + 16) * 3 + 0], fv1 * a1);
    atomicAdd(&aggv[(size_t)dn * 96 + (col + 16) * 3 + 1], fv1 * a2);
    atomicAdd(&aggv[(size_t)dn * 96 + (col + 16) * 3 + 2], fv1 * a3);
  }
}

// ---------------------------------------------------------------------------
// K3: lin2 FCTPs (scalar + 3 vector comps share B), skip blend, graph scatter
// ---------------------------------------------------------------------------
__global__ void __launch_bounds__(32)
k_fctp2(const float* __restrict__ aggs, const float* __restrict__ aggv,
        const float* __restrict__ a, const float* __restrict__ ssc,
        const v16h* __restrict__ wl2sp, const v16h* __restrict__ wl2vp,
        const int* __restrict__ batch, float* __restrict__ gsum, int N) {
  __shared__ float as2[16 * EM];
  __shared__ float ags[16 * EM];
  __shared__ float agv[16 * 96];
  const int base = blockIdx.x * 16;
  const int lane = threadIdx.x;
  const int row = lane & 15, col = lane & 15;
  const int hi = lane >> 4;
#pragma unroll
  for (int i = 0; i < 4; ++i) {
    int f4 = lane + 32 * i;
    int n = base + (f4 >> 3), c4 = (f4 & 7) * 4;
    float4 va = make_float4(0.f, 0.f, 0.f, 0.f), vs = va;
    if (n < N) {
      va = *(const float4*)(a + (size_t)n * EM + c4);
      vs = *(const float4*)(aggs + (size_t)n * EM + c4);
    }
    *(float4*)(as2 + f4 * 4) = va;
    *(float4*)(ags + f4 * 4) = vs;
  }
#pragma unroll
  for (int i = 0; i < 12; ++i) {
    int f4 = lane + 32 * i;            // 16 rows * 24 float4 per row
    int n = base + f4 / 24, c4 = (f4 % 24) * 4;
    float4 vv = make_float4(0.f, 0.f, 0.f, 0.f);
    if (n < N) vv = *(const float4*)(aggv + (size_t)n * 96 + c4);
    *(float4*)(agv + f4 * 4) = vv;
  }
  __syncthreads();
  v16h av16;
#pragma unroll
  for (int i = 0; i < 16; ++i) av16[i] = (_Float16)as2[row * EM + jmapA(i, hi)];

  const float inv_nn = 0.25f;  // 1/sqrt(16)
  v8f accS = {}, accX = {}, accY = {}, accZ = {};
#pragma unroll 2
  for (int c = 0; c < 32; ++c) {
    _Float16 xs = (_Float16)(ags[row * EM + c] * inv_nn);
    _Float16 xx = (_Float16)(agv[row * 96 + c * 3 + 0] * inv_nn);
    _Float16 xy = (_Float16)(agv[row * 96 + c * 3 + 1] * inv_nn);
    _Float16 xz = (_Float16)(agv[row * 96 + c * 3 + 2] * inv_nn);
    v16h As = av16 * xs;
    v16h Ax = av16 * xx;
    v16h Ay = av16 * xy;
    v16h Az = av16 * xz;
    v16h Bs = load_bp(wl2sp, 16, 32 * c, col, hi);
    v16h Bv = load_bp(wl2vp, 16, 32 * c, col, hi);
    accS = wmma_f16(As, Bs, accS);
    accX = wmma_f16(Ax, Bv, accX);
    accY = wmma_f16(Ay, Bv, accY);
    accZ = wmma_f16(Az, Bv, accZ);
  }
  const float inv = 1.0f / 32.0f;
  const float c_s = 0.3826834323650898f;  // sin(pi/8)
  const float c_x = 0.9238795325112867f;  // cos(pi/8)
#pragma unroll
  for (int j = 0; j < 8; ++j) {
    int M = j + (hi ? 8 : 0);
    int n = base + M;
    if (n >= N) continue;
    int g = batch[n];
    float outs = c_s * ssc[(size_t)n * OUT_MUL + col] + c_x * accS[j] * inv;
    atomicAdd(&gsum[g * 64 + col], outs);
    atomicAdd(&gsum[g * 64 + 16 + col * 3 + 0], accX[j] * inv);
    atomicAdd(&gsum[g * 64 + 16 + col * 3 + 1], accY[j] * inv);
    atomicAdd(&gsum[g * 64 + 16 + col * 3 + 2], accZ[j] * inv);
  }
}

// ---------------------------------------------------------------------------
// K4: scatter-mean finalize
// ---------------------------------------------------------------------------
__global__ void k_final(const float* __restrict__ gsum, const float* __restrict__ cnt,
                        float* __restrict__ out, int total) {
  int i = blockIdx.x * blockDim.x + threadIdx.x;
  if (i >= total) return;
  out[i] = gsum[i] / fmaxf(cnt[i >> 6], 1.0f);
}

// ---------------------------------------------------------------------------
extern "C" void kernel_launch(void* const* d_in, const int* in_sizes, int n_in,
                              void* d_out, int out_size, void* d_ws, size_t ws_size,
                              hipStream_t stream) {
  (void)n_in; (void)ws_size;
  const float* node_input = (const float*)d_in[0];
  const float* node_attr  = (const float*)d_in[1];
  const int*   edge_index = (const int*)d_in[2];
  const float* edge_vec   = (const float*)d_in[3];
  const int*   batch      = (const int*)d_in[4];
  const float* W_em   = (const float*)d_in[5];
  const float* b_em   = (const float*)d_in[6];
  const float* W_sc   = (const float*)d_in[7];
  const float* W_lin1 = (const float*)d_in[8];
  const float* W_fc1  = (const float*)d_in[9];
  const float* W_fc2  = (const float*)d_in[10];
  const float* W_l2s  = (const float*)d_in[11];
  const float* W_l2v  = (const float*)d_in[12];

  const int N = in_sizes[0] / IN_DIM;
  const int E = in_sizes[3] / 3;
  const int G = out_size / 64;

  char* ws = (char*)d_ws;
  size_t off = 0;
  auto take = [&](size_t bytes) { size_t o = off; off += (bytes + 255) & ~(size_t)255; return o; };
  size_t o_x0   = take((size_t)N * EM * 4);
  size_t o_a    = take((size_t)N * EM * 4);
  size_t o_x1   = take((size_t)N * EM * 4);
  size_t o_ssc  = take((size_t)N * OUT_MUL * 4);
  size_t o_aggs = take((size_t)N * EM * 4);
  size_t o_aggv = take((size_t)N * 96 * 4);
  size_t o_gsum = take((size_t)G * 64 * 4);
  size_t o_cnt  = take((size_t)G * 4);
  size_t o_wl1  = take((size_t)1024 * 32 * 2);
  size_t o_wsc  = take((size_t)1024 * 16 * 2);
  size_t o_wl2s = take((size_t)1024 * 16 * 2);
  size_t o_wl2v = take((size_t)1024 * 16 * 2);
  size_t o_wfc1 = take((size_t)32 * HIDP * 2);
  size_t o_wfc2 = take((size_t)128 * 64 * 2);

  float* x0   = (float*)(ws + o_x0);
  float* a    = (float*)(ws + o_a);
  float* x1   = (float*)(ws + o_x1);
  float* ssc  = (float*)(ws + o_ssc);
  float* aggs = (float*)(ws + o_aggs);
  float* aggv = (float*)(ws + o_aggv);
  float* gsum = (float*)(ws + o_gsum);
  float* cnt  = (float*)(ws + o_cnt);
  _Float16* wl1h  = (_Float16*)(ws + o_wl1);
  _Float16* wsch  = (_Float16*)(ws + o_wsc);
  _Float16* wl2sh = (_Float16*)(ws + o_wl2s);
  _Float16* wl2vh = (_Float16*)(ws + o_wl2v);
  _Float16* wfc1h = (_Float16*)(ws + o_wfc1);
  _Float16* wfc2h = (_Float16*)(ws + o_wfc2);

  // zero the atomic accumulators (aggs..cnt are contiguous in ws)
  hipMemsetAsync(ws + o_aggs, 0, o_wl1 - o_aggs, stream);

  // weight conversion + B-layout pre-swizzle
  {
    int total = 1024 * 32 + 3 * 1024 * 16 + 32 * HIDP + 128 * 64;
    k_convert<<<(total + 255) / 256, 256, 0, stream>>>(
        W_lin1, W_sc, W_l2s, W_l2v, W_fc1, W_fc2,
        wl1h, wsch, wl2sh, wl2vh, wfc1h, wfc2h);
  }
  // embeddings + graph counts
  k_embed<<<((size_t)N * EM + 255) / 256, 256, 0, stream>>>(
      node_input, node_attr, W_em, b_em, batch, x0, a, cnt, N);
  // node FCTP 1 (x1, s_sc)
  k_fctp1<<<(N + 15) / 16, 32, 0, stream>>>(x0, a, (const v16h*)wl1h,
                                            (const v16h*)wsch, x1, ssc, N);
  // edge pipeline: geometry + radial MLP + message + scatter
  k_edge<<<(E + 15) / 16, 32, 0, stream>>>(edge_index, edge_vec, x1,
                                           (const v16h*)wfc1h, (const v16h*)wfc2h,
                                           aggs, aggv, E);
  // node FCTP 2 + skip blend + per-graph accumulation
  k_fctp2<<<(N + 15) / 16, 32, 0, stream>>>(aggs, aggv, a, ssc,
                                            (const v16h*)wl2sh, (const v16h*)wl2vh,
                                            batch, gsum, N);
  // finalize mean
  k_final<<<(out_size + 255) / 256, 256, 0, stream>>>(gsum, cnt, (float*)d_out, out_size);
}
